// HGCN_23476291240115
// MI455X (gfx1250) — matrix operand order repaired and checked by reading, not compiled, not run on hardware
//
#include <hip/hip_runtime.h>
#include <hip/hip_bf16.h>

typedef __bf16 bf16_t;
typedef __attribute__((ext_vector_type(16))) __bf16 v16bf;
typedef __attribute__((ext_vector_type(8)))  float  v8f;

#define NB  64
#define NN  4096
#define NE  4096
#define NZ  32768
#define CC  256
#define MT  (NB * NN)           // 262144 total rows

// ---------------------------------------------------------------------------
// Pack W (fp32 [256 x 256], row-major K x N) into WMMA B-fragment order,
// split into bf16 hi/lo.  Fragment f = nt*8 + kt; lane holds column
// n = nt*16 + (lane&15); its 16 bf16 elements j are K = kt*32 + 16*(lane>>4) + j.
// Stored so lane's 16 values are contiguous: pack[(f*32 + lane)*16 + j].
// ---------------------------------------------------------------------------
__global__ void pack_w_kernel(const float* __restrict__ W,
                              bf16_t* __restrict__ Wh,
                              bf16_t* __restrict__ Wl) {
    int t = blockIdx.x * blockDim.x + threadIdx.x;   // 0 .. 128*32-1
    if (t >= 128 * 32) return;
    int f    = t >> 5;
    int lane = t & 31;
    int nt = f >> 3, kt = f & 7;
    int n  = nt * 16 + (lane & 15);
    int kb = kt * 32 + 16 * (lane >> 4);
#pragma unroll
    for (int j = 0; j < 16; ++j) {
        float w = W[(size_t)(kb + j) * CC + n];
        __bf16 hi = (__bf16)w;
        __bf16 lo = (__bf16)(w - (float)hi);
        Wh[(size_t)t * 16 + j] = hi;
        Wl[(size_t)t * 16 + j] = lo;
    }
}

// ---------------------------------------------------------------------------
// bf16x3 split-precision GEMM: Y[M x 256] = X[M x 256] @ W, f32 accumulate.
// 256 threads = 8 waves; wave w owns rows [blk*128 + w*16, +16).
// ---------------------------------------------------------------------------
__global__ __launch_bounds__(256)
void gemm_bf16x3_kernel(const float* __restrict__ X,
                        const bf16_t* __restrict__ Wh,
                        const bf16_t* __restrict__ Wl,
                        float* __restrict__ Y) {
    const int lane  = threadIdx.x & 31;
    const int wave  = threadIdx.x >> 5;
    const int mbase = blockIdx.x * 128 + wave * 16;
    const int row   = lane & 15;
    const int h     = lane >> 4;

    // Load this wave's 16x256 A-strip as bf16 hi/lo fragments (k = kt*32 + 16*h + j, same order as packed B).
    v16bf Ah[8], Al[8];
    const float* xrow = X + (size_t)(mbase + row) * CC;
#pragma unroll
    for (int kt = 0; kt < 8; ++kt) {
        const float4 a0 = *(const float4*)(xrow + kt * 32 + 16 * h);
        const float4 a1 = *(const float4*)(xrow + kt * 32 + 16 * h + 4);
        const float4 a2 = *(const float4*)(xrow + kt * 32 + 16 * h + 8);
        const float4 a3 = *(const float4*)(xrow + kt * 32 + 16 * h + 12);
        float v[16] = {a0.x, a0.y, a0.z, a0.w, a1.x, a1.y, a1.z, a1.w,
                       a2.x, a2.y, a2.z, a2.w, a3.x, a3.y, a3.z, a3.w};
#pragma unroll
        for (int j = 0; j < 16; ++j) {
            __bf16 hi = (__bf16)v[j];
            Ah[kt][j] = hi;
            Al[kt][j] = (__bf16)(v[j] - (float)hi);
        }
    }

    // 16 N-tiles of 16 columns each.
#pragma unroll 1
    for (int nt = 0; nt < 16; ++nt) {
        v8f acc = {};
#pragma unroll
        for (int kt = 0; kt < 8; ++kt) {
            const size_t f = ((size_t)(nt * 8 + kt) * 32 + lane) * 16;
            v16bf bh = *(const v16bf*)(Wh + f);
            v16bf bl = *(const v16bf*)(Wl + f);
            acc = __builtin_amdgcn_wmma_f32_16x16x32_bf16(
                      false, Ah[kt], false, bh, (short)0, acc, false, false);
            acc = __builtin_amdgcn_wmma_f32_16x16x32_bf16(
                      false, Al[kt], false, bh, (short)0, acc, false, false);
            acc = __builtin_amdgcn_wmma_f32_16x16x32_bf16(
                      false, Ah[kt], false, bl, (short)0, acc, false, false);
        }
        // C layout: VGPR r -> M = r + 8*h, N = lane&15.
        float* yb = Y + (size_t)(mbase + 8 * h) * CC + nt * 16 + row;
#pragma unroll
        for (int r = 0; r < 8; ++r) yb[(size_t)r * CC] = acc[r];
    }
}

// ---------------------------------------------------------------------------
__global__ void fill0_kernel(float4* __restrict__ p, unsigned n4) {
    unsigned i = blockIdx.x * blockDim.x + threadIdx.x;
    if (i < n4) p[i] = make_float4(0.f, 0.f, 0.f, 0.f);
}

__global__ void deg_count_kernel(const int* __restrict__ src,
                                 const int* __restrict__ dst,
                                 float* __restrict__ Ddeg,
                                 float* __restrict__ Bdeg) {
    int i = blockIdx.x * blockDim.x + threadIdx.x;
    if (i < NZ) {
        atomicAdd(&Ddeg[src[i]], 1.0f);
        atomicAdd(&Bdeg[dst[i]], 1.0f);
    }
}

__global__ void invert_kernel(float* __restrict__ a, int n) {
    int i = blockIdx.x * blockDim.x + threadIdx.x;
    if (i < n) { float v = a[i]; a[i] = (v > 0.f) ? 1.0f / v : 0.0f; }
}

// node -> hyperedge:  EDGE[b, dst[i], :] += XW[b, src[i], :]
__global__ __launch_bounds__(256)
void scatter_ne_kernel(const float* __restrict__ XW,
                       const int* __restrict__ src,
                       const int* __restrict__ dst,
                       float* __restrict__ EDGE) {
    const int i = blockIdx.x;                       // nnz index
    const int t = threadIdx.x;
    const int b = blockIdx.y * 4 + (t >> 6);
    const int c = (t & 63) * 4;
    const int s = src[i], d = dst[i];
    const float4 v = *(const float4*)(XW + ((size_t)b * NN + s) * CC + c);
    float* o = EDGE + ((size_t)b * NE + d) * CC + c;
    atomicAdd(o + 0, v.x); atomicAdd(o + 1, v.y);
    atomicAdd(o + 2, v.z); atomicAdd(o + 3, v.w);
}

// hyperedge -> node:  NODE[b, src[i], :] += Binv[dst[i]] * EDGE[b, dst[i], :]
__global__ __launch_bounds__(256)
void scatter_en_kernel(const float* __restrict__ EDGE,
                       const int* __restrict__ src,
                       const int* __restrict__ dst,
                       const float* __restrict__ Binv,
                       float* __restrict__ NODE) {
    const int i = blockIdx.x;
    const int t = threadIdx.x;
    const int b = blockIdx.y * 4 + (t >> 6);
    const int c = (t & 63) * 4;
    const int s = src[i], d = dst[i];
    const float sc = Binv[d];
    const float4 v = *(const float4*)(EDGE + ((size_t)b * NE + d) * CC + c);
    float* o = NODE + ((size_t)b * NN + s) * CC + c;
    atomicAdd(o + 0, v.x * sc); atomicAdd(o + 1, v.y * sc);
    atomicAdd(o + 2, v.z * sc); atomicAdd(o + 3, v.w * sc);
}

// Y[b,n,c] = Y[b,n,c] * Dinv[n] + bias[c]   (in place, float4 over C)
__global__ __launch_bounds__(256)
void finalize_kernel(float* __restrict__ Y,
                     const float* __restrict__ Dinv,
                     const float* __restrict__ bias) {
    unsigned idx = blockIdx.x * blockDim.x + threadIdx.x;  // over NB*NN*64
    if (idx >= (unsigned)NB * NN * (CC / 4)) return;
    const int c4 = idx & 63;
    const int n  = (idx >> 6) & (NN - 1);
    float4 v = ((float4*)Y)[idx];
    const float di = Dinv[n];
    const float4 bb = ((const float4*)bias)[c4];
    v.x = v.x * di + bb.x;  v.y = v.y * di + bb.y;
    v.z = v.z * di + bb.z;  v.w = v.w * di + bb.w;
    ((float4*)Y)[idx] = v;
}

// ---------------------------------------------------------------------------
extern "C" void kernel_launch(void* const* d_in, const int* in_sizes, int n_in,
                              void* d_out, int out_size, void* d_ws, size_t ws_size,
                              hipStream_t stream) {
    const float* x   = (const float*)d_in[0];
    const int*   hei = (const int*)d_in[1];
    const int*   src = hei;          // hyperedge_index[0]
    const int*   dst = hei + NZ;     // hyperedge_index[1]
    const float* W1  = (const float*)d_in[2];
    const float* b1  = (const float*)d_in[3];
    const float* W2  = (const float*)d_in[4];
    const float* b2  = (const float*)d_in[5];
    float* out = (float*)d_out;

    // Workspace layout (floats): XW | EDGE | H | Dinv | Binv | packed weights
    const size_t TEN = (size_t)NB * NN * CC;         // 67,108,864 elements
    float* XW   = (float*)d_ws;
    float* EDGE = XW   + TEN;
    float* H    = EDGE + TEN;
    float* Dinv = H    + TEN;
    float* Binv = Dinv + NN;
    bf16_t* W1h = (bf16_t*)(Binv + NE);
    bf16_t* W1l = W1h + CC * CC;
    bf16_t* W2h = W1l + CC * CC;
    bf16_t* W2l = W2h + CC * CC;

    const unsigned ten4 = (unsigned)(TEN / 4);       // float4 count = 16,777,216
    const dim3 sgrid(NZ, NB / 4);                    // scatter grid
    const int fin_blocks = (int)((TEN / 4 + 255) / 256);

    // --- prep: pack weights, degrees ---
    pack_w_kernel<<<16, 256, 0, stream>>>(W1, W1h, W1l);
    pack_w_kernel<<<16, 256, 0, stream>>>(W2, W2h, W2l);
    fill0_kernel<<<(NN + NE) / 4 / 256, 256, 0, stream>>>((float4*)Dinv, (NN + NE) / 4);
    deg_count_kernel<<<NZ / 256, 256, 0, stream>>>(src, dst, Dinv, Binv);
    invert_kernel<<<(NN + NE) / 256, 256, 0, stream>>>(Dinv, NN + NE);

    // --- layer 1: x -> H ---
    gemm_bf16x3_kernel<<<MT / 128, 256, 0, stream>>>(x, W1h, W1l, XW);
    fill0_kernel<<<ten4 / 256, 256, 0, stream>>>((float4*)EDGE, ten4);
    scatter_ne_kernel<<<sgrid, 256, 0, stream>>>(XW, src, dst, EDGE);
    fill0_kernel<<<ten4 / 256, 256, 0, stream>>>((float4*)H, ten4);
    scatter_en_kernel<<<sgrid, 256, 0, stream>>>(EDGE, src, dst, Binv, H);
    finalize_kernel<<<fin_blocks, 256, 0, stream>>>(H, Dinv, b1);

    // --- layer 2: H -> out ---
    gemm_bf16x3_kernel<<<MT / 128, 256, 0, stream>>>(H, W2h, W2l, XW);
    fill0_kernel<<<ten4 / 256, 256, 0, stream>>>((float4*)EDGE, ten4);
    scatter_ne_kernel<<<sgrid, 256, 0, stream>>>(XW, src, dst, EDGE);
    fill0_kernel<<<ten4 / 256, 256, 0, stream>>>((float4*)out, ten4);
    scatter_en_kernel<<<sgrid, 256, 0, stream>>>(EDGE, src, dst, Binv, out);
    finalize_kernel<<<fin_blocks, 256, 0, stream>>>(out, Dinv, b2);
}
